// GlossFreeVQ_42150809043139
// MI455X (gfx1250) — matrix operand order repaired and checked by reading, not compiled, not run on hardware
//
#include <hip/hip_runtime.h>
#include <hip/hip_bf16.h>

typedef _Float16 v16h __attribute__((ext_vector_type(16)));
typedef _Float16 h8   __attribute__((ext_vector_type(8)));
typedef float    v8f  __attribute__((ext_vector_type(8)));

union V16H { v16h v; h8 p[2]; };

// ---------------------------------------------------------------------------
// WMMA fragment loaders (CDNA5 16-bit layouts, wave32)
// A: 16x32 (MxK), row-major A (M x K): lane l -> m = l&15, khalf = 8*(l>>4)
//    halves 0..7  = K: khalf+0..7        (16B)
//    halves 8..15 = K: 16+khalf+0..7     (16B)
// B: 32x16 (KxN) fed from Bt (N x K) row-major: lane l -> n = l&15,
//    halves 0..7 = K: 8*(l>>4)+0..7, halves 8..15 = K: 16+8*(l>>4)+0..7 (two 16B loads)
// C/D: vgpr r -> row r + 8*(l>>4), col l&15
// ---------------------------------------------------------------------------
__device__ __forceinline__ v16h load_a_frag(const _Float16* p, int h) {
    V16H r;
    r.p[0] = *(const h8*)(p + 8 * h);
    r.p[1] = *(const h8*)(p + 16 + 8 * h);
    return r.v;
}
__device__ __forceinline__ v16h load_b_frag(const _Float16* p, int h) {
    V16H r;
    r.p[0] = *(const h8*)(p + 8 * h);
    r.p[1] = *(const h8*)(p + 16 + 8 * h);
    return r.v;
}
__device__ __forceinline__ v8f wmma16(v16h a, v16h b, v8f c) {
    return __builtin_amdgcn_wmma_f32_16x16x32_f16(false, a, false, b, (short)0, c,
                                                  false, false);
}

// ---------------------------------------------------------------------------
// small utility kernels
// ---------------------------------------------------------------------------
__global__ void k_zero_acc(float* acc) {
    if (threadIdx.x < 8) acc[threadIdx.x] = 0.0f;
}

__global__ void k_f32_to_f16(const float* __restrict__ in, _Float16* __restrict__ out,
                             int n) {
    int i = blockIdx.x * 256 + threadIdx.x;
    if (i < n) out[i] = (_Float16)in[i];
}

// in: R x C row-major f32 -> out: C x R row-major f16 (Bt layout)
__global__ void k_transpose_f16(const float* __restrict__ in, _Float16* __restrict__ out,
                                int R, int C) {
    int i = blockIdx.x * 256 + threadIdx.x;
    if (i < R * C) {
        int r = i / C, c = i - r * C;
        out[c * R + r] = (_Float16)in[i];
    }
}

// per-code squared norm + f16 conversion of the codebook (already Bt layout)
__global__ void k_codebook_prep(const float* __restrict__ cb, _Float16* __restrict__ cbh,
                                float* __restrict__ cnorm) {
    __shared__ float red[256];
    int n = blockIdx.x, tid = threadIdx.x;
    float s = 0.0f;
    for (int c = tid; c < 512; c += 256) {
        float v = cb[n * 512 + c];
        cbh[n * 512 + c] = (_Float16)v;
        s += v * v;
    }
    red[tid] = s;
    __syncthreads();
    for (int off = 128; off; off >>= 1) {
        if (tid < off) red[tid] += red[tid + off];
        __syncthreads();
    }
    if (tid == 0) cnorm[n] = red[0];
}

// ---------------------------------------------------------------------------
// GEMM + bias + LayerNorm + ReLU  (workgroup = 16 rows x full N, 8 waves)
// ---------------------------------------------------------------------------
template <int N, int K, int NPER>
__global__ __launch_bounds__(256) void k_gemm_ln_relu(
    const _Float16* __restrict__ A, const _Float16* __restrict__ Bt,
    const float* __restrict__ bias, const float* __restrict__ g,
    const float* __restrict__ be, _Float16* __restrict__ outH,
    float* __restrict__ outF) {
    __shared__ __align__(16) float S[16 * N];
    __shared__ float psum[256], psq[256], mu_s[16], rs_s[16];
    const int tid = threadIdx.x, lane = tid & 31, w = tid >> 5;
    const int h = lane >> 4, ln16 = lane & 15;
    const int m0 = blockIdx.x * 16;
    const int arow = m0 + ln16;

    v8f acc[NPER];
#pragma unroll
    for (int i = 0; i < NPER; i++) acc[i] = {};

    for (int k0 = 0; k0 < K; k0 += 32) {
        v16h a = load_a_frag(A + arow * K + k0, h);
#pragma unroll
        for (int i = 0; i < NPER; i++) {
            int nt = w + 8 * i;
            v16h b = load_b_frag(Bt + (nt * 16 + ln16) * K + k0, h);
            acc[i] = wmma16(a, b, acc[i]);
        }
    }
#pragma unroll
    for (int i = 0; i < NPER; i++) {
        int col = (w + 8 * i) * 16 + ln16;
#pragma unroll
        for (int r = 0; r < 8; r++)
            S[(r + 8 * h) * N + col] = acc[i][r] + bias[col];
    }
    __syncthreads();
    {
        int rrow = tid >> 4, c0 = tid & 15;
        float s = 0.0f, q = 0.0f;
        for (int c = c0; c < N; c += 16) {
            float v = S[rrow * N + c];
            s += v;
            q += v * v;
        }
        psum[tid] = s;
        psq[tid] = q;
    }
    __syncthreads();
    if (tid < 16) {
        float s = 0.0f, q = 0.0f;
        for (int j = 0; j < 16; j++) { s += psum[tid * 16 + j]; q += psq[tid * 16 + j]; }
        float mu = s / (float)N;
        float var = q / (float)N - mu * mu;
        mu_s[tid] = mu;
        rs_s[tid] = rsqrtf(var + 1e-5f);
    }
    __syncthreads();
    for (int idx = tid; idx < 16 * N; idx += 256) {
        int rrow = idx / N, col = idx - rrow * N;
        float v = (S[idx] - mu_s[rrow]) * rs_s[rrow] * g[col] + be[col];
        v = fmaxf(v, 0.0f);
        int o = (m0 + rrow) * N + col;
        outH[o] = (_Float16)v;
        if (outF) outF[o] = v;
    }
}

// ---------------------------------------------------------------------------
// GEMM + bias + ReLU (decoder layer 1)
// ---------------------------------------------------------------------------
template <int N, int K, int NPER>
__global__ __launch_bounds__(256) void k_gemm_bias_relu(
    const _Float16* __restrict__ A, const _Float16* __restrict__ Bt,
    const float* __restrict__ bias, _Float16* __restrict__ outH) {
    const int tid = threadIdx.x, lane = tid & 31, w = tid >> 5;
    const int h = lane >> 4, ln16 = lane & 15;
    const int m0 = blockIdx.x * 16;
    const int arow = m0 + ln16;

    v8f acc[NPER];
#pragma unroll
    for (int i = 0; i < NPER; i++) acc[i] = {};
    for (int k0 = 0; k0 < K; k0 += 32) {
        v16h a = load_a_frag(A + arow * K + k0, h);
#pragma unroll
        for (int i = 0; i < NPER; i++) {
            int nt = w + 8 * i;
            v16h b = load_b_frag(Bt + (nt * 16 + ln16) * K + k0, h);
            acc[i] = wmma16(a, b, acc[i]);
        }
    }
#pragma unroll
    for (int i = 0; i < NPER; i++) {
        int col = (w + 8 * i) * 16 + ln16;
#pragma unroll
        for (int r = 0; r < 8; r++) {
            int row = m0 + r + 8 * h;
            float v = fmaxf(acc[i][r] + bias[col], 0.0f);
            outH[row * N + col] = (_Float16)v;
        }
    }
}

// ---------------------------------------------------------------------------
// GEMM + bias, fused reconstruction-loss reduction (decoder layer 2)
// ---------------------------------------------------------------------------
template <int N, int K, int NPER>
__global__ __launch_bounds__(256) void k_gemm_recon(
    const _Float16* __restrict__ A, const _Float16* __restrict__ Bt,
    const float* __restrict__ bias, const float* __restrict__ X,
    float* __restrict__ recon_acc) {
    __shared__ float red[256];
    const int tid = threadIdx.x, lane = tid & 31, w = tid >> 5;
    const int h = lane >> 4, ln16 = lane & 15;
    const int m0 = blockIdx.x * 16;
    const int arow = m0 + ln16;

    v8f acc[NPER];
#pragma unroll
    for (int i = 0; i < NPER; i++) acc[i] = {};
    for (int k0 = 0; k0 < K; k0 += 32) {
        v16h a = load_a_frag(A + arow * K + k0, h);
#pragma unroll
        for (int i = 0; i < NPER; i++) {
            int nt = w + 8 * i;
            v16h b = load_b_frag(Bt + (nt * 16 + ln16) * K + k0, h);
            acc[i] = wmma16(a, b, acc[i]);
        }
    }
    float local = 0.0f;
#pragma unroll
    for (int i = 0; i < NPER; i++) {
        int col = (w + 8 * i) * 16 + ln16;
#pragma unroll
        for (int r = 0; r < 8; r++) {
            int row = m0 + r + 8 * h;
            float v = acc[i][r] + bias[col];
            float d = v - X[row * N + col];
            local += d * d;
        }
    }
    red[tid] = local;
    __syncthreads();
    for (int off = 128; off; off >>= 1) {
        if (tid < off) red[tid] += red[tid + off];
        __syncthreads();
    }
    if (tid == 0) atomicAdd(recon_acc, red[0]);
}

// ---------------------------------------------------------------------------
// Distance argmin over 8192 codes + gather quantized + commitment loss.
// score = |c|^2 - 2 e.c  (|e|^2 constant per row, irrelevant for argmin)
// ---------------------------------------------------------------------------
__global__ __launch_bounds__(256) void k_argmin_quant(
    const _Float16* __restrict__ encH, const float* __restrict__ encF,
    const _Float16* __restrict__ cbH, const float* __restrict__ cnorm,
    const float* __restrict__ cbF, float* __restrict__ idx_out,
    float* __restrict__ qF, _Float16* __restrict__ qH,
    float* __restrict__ commit_acc) {
    __shared__ __align__(16) _Float16 Atile[16 * 512];
    __shared__ float cv[16 * 128];
    __shared__ int ci[16 * 128];
    __shared__ int sidx[16];
    __shared__ float red[256];
    const int tid = threadIdx.x, lane = tid & 31, w = tid >> 5;
    const int h = lane >> 4, ln16 = lane & 15;
    const int m0 = blockIdx.x * 16;

    // stage 16x512 f16 encoded strip into LDS (16 KB)
    for (int i = tid; i < 16 * 512 / 8; i += 256)
        ((h8*)Atile)[i] = ((const h8*)(encH + m0 * 512))[i];
    __syncthreads();

    float best[8];
    int bidx[8];
#pragma unroll
    for (int r = 0; r < 8; r++) { best[r] = 3.4e38f; bidx[r] = 0; }

    for (int nt = w; nt < 512; nt += 8) {
        v8f acc = {};
        const _Float16* brow = cbH + (nt * 16 + ln16) * 512;
#pragma unroll
        for (int k0 = 0; k0 < 512; k0 += 32) {
            v16h a = load_a_frag(Atile + ln16 * 512 + k0, h);
            v16h b = load_b_frag(brow + k0, h);
            acc = wmma16(a, b, acc);
        }
        int n = nt * 16 + ln16;
        float cn = cnorm[n];
#pragma unroll
        for (int r = 0; r < 8; r++) {
            float score = cn - 2.0f * acc[r];
            if (score < best[r]) { best[r] = score; bidx[r] = n; }
        }
    }
#pragma unroll
    for (int r = 0; r < 8; r++) {
        int rw = r + 8 * h;
        int slot = w * 16 + ln16;
        cv[rw * 128 + slot] = best[r];
        ci[rw * 128 + slot] = bidx[r];
    }
    __syncthreads();
    if (tid < 16) {
        float bv = 3.4e38f;
        int bi = 0x7fffffff;
        for (int j = 0; j < 128; j++) {
            float v = cv[tid * 128 + j];
            int id = ci[tid * 128 + j];
            if (v < bv || (v == bv && id < bi)) { bv = v; bi = id; }
        }
        sidx[tid] = bi;
        idx_out[m0 + tid] = (float)bi;  // indices flattened (B,T)
    }
    __syncthreads();
    float local = 0.0f;
    for (int i = tid; i < 16 * 512; i += 256) {
        int rw = i >> 9, c = i & 511;
        float q = cbF[sidx[rw] * 512 + c];
        int o = (m0 + rw) * 512 + c;
        qF[o] = q;
        qH[o] = (_Float16)q;
        float d = encF[o] - q;
        local += d * d;
    }
    red[tid] = local;
    __syncthreads();
    for (int off = 128; off; off >>= 1) {
        if (tid < off) red[tid] += red[tid + off];
        __syncthreads();
    }
    if (tid == 0) atomicAdd(commit_acc, red[0]);
}

// ---------------------------------------------------------------------------
// pooled[b, :] = mean_t quantized[b, t, :]
// ---------------------------------------------------------------------------
__global__ void k_pool(const float* __restrict__ qF, float* __restrict__ pooled) {
    int b = blockIdx.x, tid = threadIdx.x;
    for (int c = tid; c < 512; c += 256) {
        float s = 0.0f;
        for (int t = 0; t < 512; t++) s += qF[(b * 512 + t) * 512 + c];
        pooled[b * 512 + c] = s * (1.0f / 512.0f);
    }
}

// ---------------------------------------------------------------------------
// projection head + contrastive loss (tiny; one workgroup)
// ---------------------------------------------------------------------------
__global__ __launch_bounds__(256) void k_contrastive(
    const float* __restrict__ pooled, const float* __restrict__ Wp1,
    const float* __restrict__ bp1, const float* __restrict__ Wp2,
    const float* __restrict__ bp2, float* __restrict__ contr_out) {
    __shared__ float P1[16 * 256];
    __shared__ float PR[16 * 128];
    __shared__ float nrm[16];
    __shared__ float sim[16 * 16];
    __shared__ float ls[16];
    int tid = threadIdx.x;
    for (int idx = tid; idx < 16 * 256; idx += 256) {
        int b = idx >> 8, j = idx & 255;
        float s = bp1[j];
        for (int c = 0; c < 512; c++) s += pooled[b * 512 + c] * Wp1[c * 256 + j];
        P1[idx] = fmaxf(s, 0.0f);
    }
    __syncthreads();
    for (int idx = tid; idx < 16 * 128; idx += 256) {
        int b = idx >> 7, j = idx & 127;
        float s = bp2[j];
        for (int c = 0; c < 256; c++) s += P1[b * 256 + c] * Wp2[c * 128 + j];
        PR[idx] = s;
    }
    __syncthreads();
    if (tid < 16) {
        float s = 0.0f;
        for (int j = 0; j < 128; j++) { float v = PR[tid * 128 + j]; s += v * v; }
        nrm[tid] = fmaxf(sqrtf(s), 1e-12f);
    }
    __syncthreads();
    for (int idx = tid; idx < 16 * 128; idx += 256) PR[idx] /= nrm[idx >> 7];
    __syncthreads();
    if (tid < 256) {
        int i = tid >> 4, j = tid & 15;
        float s = 0.0f;
        for (int c = 0; c < 128; c++) s += PR[i * 128 + c] * PR[j * 128 + c];
        sim[tid] = s * 10.0f;  // 1/TEMP
    }
    __syncthreads();
    if (tid < 16) {
        float mx = -3.4e38f;
        for (int j = 0; j < 16; j++) mx = fmaxf(mx, sim[tid * 16 + j]);
        float s = 0.0f;
        for (int j = 0; j < 16; j++) s += expf(sim[tid * 16 + j] - mx);
        ls[tid] = mx + logf(s) - sim[tid * 16 + tid];
    }
    __syncthreads();
    if (tid == 0) {
        float s = 0.0f;
        for (int j = 0; j < 16; j++) s += ls[j];
        contr_out[0] = s * (1.0f / 16.0f);
    }
}

__global__ void k_finalize(const float* __restrict__ acc, float* __restrict__ out) {
    if (threadIdx.x == 0) {
        const float inv = 1.0f / (8192.0f * 512.0f);
        float recon = acc[0] * inv;
        float commit = acc[1] * inv;
        float codeb = commit;  // identical forward value
        float contr = acc[2];
        out[0] = recon;
        out[1] = commit;
        out[2] = codeb;
        out[3] = contr;
        out[4] = recon * 1.0f + commit * 0.25f + codeb * 1.0f + contr * 0.1f;
    }
}

// ---------------------------------------------------------------------------
extern "C" void kernel_launch(void* const* d_in, const int* in_sizes, int n_in,
                              void* d_out, int out_size, void* d_ws, size_t ws_size,
                              hipStream_t stream) {
    const float* x   = (const float*)d_in[0];
    const float* W1  = (const float*)d_in[1];
    const float* b1  = (const float*)d_in[2];
    const float* g1  = (const float*)d_in[3];
    const float* be1 = (const float*)d_in[4];
    const float* W2  = (const float*)d_in[5];
    const float* b2  = (const float*)d_in[6];
    const float* g2  = (const float*)d_in[7];
    const float* be2 = (const float*)d_in[8];
    const float* cb  = (const float*)d_in[9];
    const float* Wd1 = (const float*)d_in[10];
    const float* bd1 = (const float*)d_in[11];
    const float* Wd2 = (const float*)d_in[12];
    const float* bd2 = (const float*)d_in[13];
    const float* Wp1 = (const float*)d_in[14];
    const float* bp1 = (const float*)d_in[15];
    const float* Wp2 = (const float*)d_in[16];
    const float* bp2 = (const float*)d_in[17];
    float* out = (float*)d_out;

    const int M = 8192, D = 512, H = 768, C = 512, KC = 8192;

    char* base = (char*)d_ws;
    size_t off = 0;
    auto take = [&](size_t bytes) -> char* {
        char* p = base + off;
        off = (off + bytes + 255) & ~(size_t)255;
        return p;
    };
    _Float16* xh    = (_Float16*)take((size_t)M * D * 2);
    _Float16* w1t   = (_Float16*)take((size_t)H * D * 2);
    _Float16* w2t   = (_Float16*)take((size_t)C * H * 2);
    _Float16* wd1t  = (_Float16*)take((size_t)H * C * 2);
    _Float16* wd2t  = (_Float16*)take((size_t)D * H * 2);
    _Float16* cbh   = (_Float16*)take((size_t)KC * C * 2);
    float*    cnorm = (float*)take((size_t)KC * 4);
    _Float16* hh    = (_Float16*)take((size_t)M * H * 2);
    _Float16* hdh   = (_Float16*)take((size_t)M * H * 2);
    float*    encF  = (float*)take((size_t)M * C * 4);
    _Float16* encH  = (_Float16*)take((size_t)M * C * 2);
    float*    qF    = (float*)take((size_t)M * C * 4);
    _Float16* qH    = (_Float16*)take((size_t)M * C * 2);
    float*    pool  = (float*)take((size_t)16 * C * 4);
    float*    accs  = (float*)take(8 * 4);

    k_zero_acc<<<1, 32, 0, stream>>>(accs);

    k_f32_to_f16<<<(M * D + 255) / 256, 256, 0, stream>>>(x, xh, M * D);
    k_transpose_f16<<<(D * H + 255) / 256, 256, 0, stream>>>(W1, w1t, D, H);
    k_transpose_f16<<<(H * C + 255) / 256, 256, 0, stream>>>(W2, w2t, H, C);
    k_transpose_f16<<<(C * H + 255) / 256, 256, 0, stream>>>(Wd1, wd1t, C, H);
    k_transpose_f16<<<(H * D + 255) / 256, 256, 0, stream>>>(Wd2, wd2t, H, D);
    k_codebook_prep<<<KC, 256, 0, stream>>>(cb, cbh, cnorm);

    // encoder
    k_gemm_ln_relu<768, 512, 6><<<M / 16, 256, 0, stream>>>(xh, w1t, b1, g1, be1, hh,
                                                            (float*)nullptr);
    k_gemm_ln_relu<512, 768, 4><<<M / 16, 256, 0, stream>>>(hh, w2t, b2, g2, be2, encH,
                                                            encF);
    // vector quantization (indices -> out[0:8192] as float)
    k_argmin_quant<<<M / 16, 256, 0, stream>>>(encH, encF, cbh, cnorm, cb, out, qF, qH,
                                               accs + 1);
    // decoder
    k_gemm_bias_relu<768, 512, 6><<<M / 16, 256, 0, stream>>>(qH, wd1t, bd1, hdh);
    k_gemm_recon<512, 768, 4><<<M / 16, 256, 0, stream>>>(hdh, wd2t, bd2, x, accs + 0);

    // contrastive head
    k_pool<<<16, 256, 0, stream>>>(qF, pool);
    k_contrastive<<<1, 256, 0, stream>>>(pool, Wp1, bp1, Wp2, bp2, accs + 2);

    k_finalize<<<1, 32, 0, stream>>>(accs, out + 8192);
}
